// STEncoderBlock_66417374265422
// MI455X (gfx1250) — matrix operand order repaired and checked by reading, not compiled, not run on hardware
//
#include <hip/hip_runtime.h>
#include <hip/hip_bf16.h>
#include <math.h>

// ---------------------------------------------------------------------------
// ST-Encoder block for MI455X (gfx1250, wave32, WMMA).
// All GEMM-shaped work runs on v_wmma_f32_16x16x32_f16 (f16 in, f32 accum).
// GEMMs are M2-tiled (each wave = 2 x 16 rows) so every LDS B-fragment feeds
// two WMMAs; spatial-attention K/V staging uses GLOBAL_LOAD_ASYNC_TO_LDS_B128.
// ---------------------------------------------------------------------------

typedef __attribute__((ext_vector_type(16))) _Float16 v16h;
typedef __attribute__((ext_vector_type(8)))  _Float16 v8h;
typedef __attribute__((ext_vector_type(8)))  float    v8f;

#define B_   4
#define T_   24
#define N_   512
#define D_   192
#define H_   4
#define HD_  16
#define BR_  64
#define HID_ 768
#define TOK  (B_ * T_ * N_)     // 49152 tokens
#define SCALE_ 0.25f            // HD^-0.5 = 16^-0.5

// A-fragment loader: 16x32 f16 A matrix, row = (lane&15).
// Per-lane 16 halves: lanes 0-15 hold k={0..15}, lanes 16-31 k={16..31}.
// p points at row start + k0 (k0 multiple of 32).  Two 16B loads.
static __device__ __forceinline__ v16h ldA16(const _Float16* p, int halfsel) {
  const _Float16* q = p + halfsel * 16;
  v8h lo = *(const v8h*)(q);
  v8h hi = *(const v8h*)(q + 8);
  v16h a;
#pragma unroll
  for (int i = 0; i < 8; ++i) { a[i] = lo[i]; a[i + 8] = hi[i]; }
  return a;
}

// ---------------------------------------------------------------------------
// LayerNorm (f32 in -> f16 out), one wave per token, D=192 = 6 per lane.
// ---------------------------------------------------------------------------
__global__ __launch_bounds__(256) void ln_f16_kernel(
    const float* __restrict__ x, const float* __restrict__ w,
    const float* __restrict__ bb, _Float16* __restrict__ out) {
  int wave = (blockIdx.x * blockDim.x + threadIdx.x) >> 5;
  int lane = threadIdx.x & 31;
  if (wave >= TOK) return;
  const float* row = x + (size_t)wave * D_;
  float v[6], s = 0.f, s2 = 0.f;
#pragma unroll
  for (int i = 0; i < 6; ++i) {
    v[i] = row[lane + 32 * i];
    s += v[i]; s2 += v[i] * v[i];
  }
#pragma unroll
  for (int off = 16; off >= 1; off >>= 1) {
    s  += __shfl_xor(s,  off, 32);
    s2 += __shfl_xor(s2, off, 32);
  }
  float mu  = s * (1.0f / D_);
  float var = s2 * (1.0f / D_) - mu * mu;
  float inv = rsqrtf(var + 1e-5f);
  _Float16* orow = out + (size_t)wave * D_;
#pragma unroll
  for (int i = 0; i < 6; ++i) {
    int c = lane + 32 * i;
    orow[c] = (_Float16)((v[i] - mu) * inv * w[c] + bb[c]);
  }
}

// ---------------------------------------------------------------------------
// QKV projection: [TOK,192] f16 @ [192,64] f32 + bias -> [TOK,64] f16.
// 8 waves/WG, each wave = 32 rows (2 M-tiles) x 64 cols (4 n-tiles).
// Weights transposed into LDS as Wt[n][k]; each B-frag feeds 2 WMMAs.
// ---------------------------------------------------------------------------
__global__ __launch_bounds__(256) void qkv_gemm_kernel(
    const _Float16* __restrict__ A, const float* __restrict__ W,
    const float* __restrict__ bias, _Float16* __restrict__ out) {
  __shared__ __align__(64) _Float16 Wt[64 * 192];
  int tid = threadIdx.x;
  for (int i = tid; i < 64 * 192; i += 256) {
    int n = i / 192, kk = i % 192;
    Wt[i] = (_Float16)W[kk * 64 + n];
  }
  __syncthreads();
  int lane = tid & 31, wid = tid >> 5;
  int halfsel = lane >> 4, lm = lane & 15;
  int rowBase = blockIdx.x * 256 + wid * 32;

  v8f z = {};
  v8f acc0[4], acc1[4];
#pragma unroll
  for (int t = 0; t < 4; ++t) { acc0[t] = z; acc1[t] = z; }

  const _Float16* Arow0 = A + (size_t)(rowBase + lm) * 192;
  const _Float16* Arow1 = Arow0 + (size_t)16 * 192;
#pragma unroll
  for (int ks = 0; ks < 6; ++ks) {
    v16h a0 = ldA16(Arow0 + ks * 32, halfsel);
    v16h a1 = ldA16(Arow1 + ks * 32, halfsel);
#pragma unroll
    for (int nt = 0; nt < 4; ++nt) {
      v16h b = *(const v16h*)(Wt + (nt * 16 + lm) * 192 + ks * 32 + halfsel * 16);
      acc0[nt] = __builtin_amdgcn_wmma_f32_16x16x32_f16(
          false, a0, false, b, (short)0, acc0[nt], false, false);
      acc1[nt] = __builtin_amdgcn_wmma_f32_16x16x32_f16(
          false, a1, false, b, (short)0, acc1[nt], false, false);
    }
  }
#pragma unroll
  for (int nt = 0; nt < 4; ++nt) {
    int col = nt * 16 + lm;
    float bs = bias[col];
#pragma unroll
    for (int r = 0; r < 8; ++r) {
      int tok0 = rowBase + halfsel * 8 + r;
      out[(size_t)tok0 * 64 + col]        = (_Float16)(acc0[nt][r] + bs);
      out[(size_t)(tok0 + 16) * 64 + col] = (_Float16)(acc1[nt][r] + bs);
    }
  }
}

// ---------------------------------------------------------------------------
// Temporal attention. The torch raw-reshape means instance (b,n') reads the
// CONTIGUOUS token block j = n'*24 + t' of the [B, T*N, 64] projections, and
// writes t_x[b, t', n', h*16+d]. One wave per (b,n'); 2048 instances.
// ---------------------------------------------------------------------------
__global__ __launch_bounds__(32) void temporal_attn_kernel(
    const _Float16* __restrict__ q, const _Float16* __restrict__ k,
    const _Float16* __restrict__ v, _Float16* __restrict__ attn_cat) {
  __shared__ _Float16 Ql[T_ * BR_], Kl[T_ * BR_], Vl[T_ * BR_];
  int b = blockIdx.x >> 9;      // / N_
  int n = blockIdx.x & (N_ - 1);
  int lane = threadIdx.x;
  size_t base = ((size_t)b * (T_ * N_) + (size_t)n * T_) * BR_;
  const unsigned int* qs = (const unsigned int*)(q + base);
  const unsigned int* ks = (const unsigned int*)(k + base);
  const unsigned int* vs = (const unsigned int*)(v + base);
  for (int i = lane; i < T_ * BR_ / 2; i += 32) {
    ((unsigned int*)Ql)[i] = qs[i];
    ((unsigned int*)Kl)[i] = ks[i];
    ((unsigned int*)Vl)[i] = vs[i];
  }
  __syncthreads();
  int t = lane;
  if (t < T_) {
#pragma unroll
    for (int h = 0; h < H_; ++h) {
      float s[T_]; float mx = -3.0e38f;
      for (int j = 0; j < T_; ++j) {
        float a = 0.f;
#pragma unroll
        for (int d = 0; d < HD_; ++d)
          a += (float)Ql[t * BR_ + h * HD_ + d] * (float)Kl[j * BR_ + h * HD_ + d];
        a *= SCALE_;
        s[j] = a; mx = fmaxf(mx, a);
      }
      float sum = 0.f;
      for (int j = 0; j < T_; ++j) { float p = expf(s[j] - mx); s[j] = p; sum += p; }
      float rinv = 1.0f / sum;
      size_t ob = ((size_t)(b * T_ + t) * N_ + n) * D_ + h * HD_;
#pragma unroll
      for (int d = 0; d < HD_; ++d) {
        float o = 0.f;
        for (int j = 0; j < T_; ++j) o += s[j] * (float)Vl[j * BR_ + h * HD_ + d];
        attn_cat[ob + d] = (_Float16)(o * rinv);
      }
    }
  }
}

// ---------------------------------------------------------------------------
// Spatial attention (geo / sem), flash-style online softmax with WMMA.
// Grid: (qblock 0..3, b*T 0..95, h 0..3). 8 waves x 16 query rows = 128 rows.
// K/V rows are staged global->LDS with GLOBAL_LOAD_ASYNC_TO_LDS_B128
// (ASYNCcnt-tracked DMA, no VGPR round trip), then V is transposed in LDS.
// Head-dim is padded to K=32 by zeroing the upper-half-lane B fragment.
// P round-trips D-layout -> A-layout via a 1 KB wave-private LDS tile
// (same-wave DS ops are in-order; no barrier needed).
// ---------------------------------------------------------------------------
__global__ __launch_bounds__(256) void spatial_attn_kernel(
    const _Float16* __restrict__ q, const _Float16* __restrict__ k,
    const _Float16* __restrict__ v, const unsigned char* __restrict__ mask,
    _Float16* __restrict__ attn_cat, int coff) {
  __shared__ __align__(64) _Float16 Kl[N_ * HD_];      // 16 KB [n][d]
  __shared__ __align__(64) _Float16 Vl[N_ * HD_];      // 16 KB [n][d] (staging)
  __shared__ __align__(64) _Float16 Vt[HD_ * N_];      // 16 KB [d][n]
  __shared__ __align__(64) _Float16 Pl[8 * 16 * 32];   //  8 KB per-wave P
  int tid = threadIdx.x;
  int bt  = blockIdx.y;
  int h   = blockIdx.z;
  size_t inbase = (size_t)bt * N_ * BR_ + h * HD_;

  // Async-stage K and V rows: 16B per lane, both sides 16B aligned.
  for (int c = tid; c < (N_ * HD_) / 8; c += 256) {
    int n = c >> 1, d8 = (c & 1) * 8;
    unsigned ldsK = (unsigned)(size_t)(&Kl[n * HD_ + d8]);
    unsigned ldsV = (unsigned)(size_t)(&Vl[n * HD_ + d8]);
    const _Float16* gk = k + inbase + (size_t)n * BR_ + d8;
    const _Float16* gv = v + inbase + (size_t)n * BR_ + d8;
    asm volatile("global_load_async_to_lds_b128 %0, %1, off"
                 :: "v"(ldsK), "v"(gk) : "memory");
    asm volatile("global_load_async_to_lds_b128 %0, %1, off"
                 :: "v"(ldsV), "v"(gv) : "memory");
  }
  asm volatile("s_wait_asynccnt 0x0" ::: "memory");
  __syncthreads();
  // Transpose V in LDS: Vt[d][n] = Vl[n][d].
  for (int i = tid; i < N_ * HD_; i += 256)
    Vt[(i & 15) * N_ + (i >> 4)] = Vl[i];
  __syncthreads();

  int lane = tid & 31, wid = tid >> 5;
  int halfsel = lane >> 4, lm = lane & 15;
  int qbase = blockIdx.x * 128 + wid * 16;

  // Q A-fragment: k>=16 of the padded head-dim are zero -> upper-half lanes zero.
  v16h aq;
  {
    const _Float16* qp = q + inbase + (size_t)(qbase + lm) * BR_;
    v8h qlo = *(const v8h*)qp;
    v8h qhi = *(const v8h*)(qp + 8);
#pragma unroll
    for (int i = 0; i < 8; ++i) { aq[i] = halfsel ? (_Float16)0 : qlo[i]; aq[i + 8] = halfsel ? (_Float16)0 : qhi[i]; }
  }

  v8f o = {};
  float m[8], l[8];
#pragma unroll
  for (int r = 0; r < 8; ++r) { m[r] = -3.0e38f; l[r] = 0.f; }
  _Float16* Pw = Pl + wid * (16 * 32);

  for (int c = 0; c < N_ / 32; ++c) {
    // S = Q @ K^T for 32 keys (two 16-wide tiles).  B-frag: element e has
    // K(=hd) = e + 16*halfsel; padded hd>=16 rows are zero for upper lanes.
    v16h bk0 = *(const v16h*)(Kl + ((c * 32)      + lm) * HD_);
    v16h bk1 = *(const v16h*)(Kl + ((c * 32 + 16) + lm) * HD_);
    if (halfsel) {
#pragma unroll
      for (int i = 0; i < 16; ++i) { bk0[i] = (_Float16)0; bk1[i] = (_Float16)0; }
    }
    v8f d0 = {}, d1 = {};
    d0 = __builtin_amdgcn_wmma_f32_16x16x32_f16(false, aq, false, bk0, (short)0, d0, false, false);
    d1 = __builtin_amdgcn_wmma_f32_16x16x32_f16(false, aq, false, bk1, (short)0, d1, false, false);

    // Scale + mask (D layout: row = r + 8*halfsel, key = 16*tile + lm).
#pragma unroll
    for (int r = 0; r < 8; ++r) {
      int qn = qbase + halfsel * 8 + r;
      int k0 = c * 32 + lm;
      float s0 = d0[r] * SCALE_, s1 = d1[r] * SCALE_;
      if (mask[(size_t)qn * N_ + k0])      s0 = -3.0e38f;
      if (mask[(size_t)qn * N_ + k0 + 16]) s1 = -3.0e38f;
      d0[r] = s0; d1[r] = s1;
    }
    // Online softmax: row values live across the 16 lanes of each half.
#pragma unroll
    for (int r = 0; r < 8; ++r) {
      float mx = fmaxf(d0[r], d1[r]);
#pragma unroll
      for (int off = 1; off < 16; off <<= 1) mx = fmaxf(mx, __shfl_xor(mx, off, 32));
      float mn    = fmaxf(m[r], mx);
      float alpha = expf(m[r] - mn);
      float p0    = expf(d0[r] - mn);
      float p1    = expf(d1[r] - mn);
      float rs    = p0 + p1;
#pragma unroll
      for (int off = 1; off < 16; off <<= 1) rs += __shfl_xor(rs, off, 32);
      l[r] = l[r] * alpha + rs;
      m[r] = mn;
      o[r] *= alpha;
      Pw[(halfsel * 8 + r) * 32 + lm]      = (_Float16)p0;
      Pw[(halfsel * 8 + r) * 32 + 16 + lm] = (_Float16)p1;
    }
    // O += P(16x32) @ V(32x16): A-frag of P from LDS, B-frag from V^T.
    v16h ap;
    {
      const _Float16* pp = Pw + lm * 32 + halfsel * 16;
      v8h plo = *(const v8h*)pp;
      v8h phi = *(const v8h*)(pp + 8);
#pragma unroll
      for (int i = 0; i < 8; ++i) { ap[i] = plo[i]; ap[i + 8] = phi[i]; }
    }
    v16h bv = *(const v16h*)(Vt + lm * N_ + c * 32 + halfsel * 16);
    o = __builtin_amdgcn_wmma_f32_16x16x32_f16(false, ap, false, bv, (short)0, o, false, false);
  }
#pragma unroll
  for (int r = 0; r < 8; ++r) {
    int qn = qbase + halfsel * 8 + r;
    float rcp = 1.0f / l[r];
    attn_cat[((size_t)bt * N_ + qn) * D_ + coff + h * HD_ + lm] = (_Float16)(o[r] * rcp);
  }
}

// ---------------------------------------------------------------------------
// proj + residual: x1 = x + attn_cat @ proj_w + proj_b (f32 out).
// Column group of 96 via blockIdx.y; 6 n-tiles x 2 M-tiles per wave.
// ---------------------------------------------------------------------------
__global__ __launch_bounds__(256) void proj_gemm_kernel(
    const _Float16* __restrict__ A, const float* __restrict__ W,
    const float* __restrict__ bias, const float* __restrict__ xres,
    float* __restrict__ x1) {
  __shared__ __align__(64) _Float16 Wt[96 * 192];
  int tid = threadIdx.x;
  int c0 = blockIdx.y * 96;
  for (int i = tid; i < 96 * 192; i += 256) {
    int n = i / 192, kk = i % 192;
    Wt[i] = (_Float16)W[kk * 192 + c0 + n];
  }
  __syncthreads();
  int lane = tid & 31, wid = tid >> 5, halfsel = lane >> 4, lm = lane & 15;
  int rowBase = blockIdx.x * 256 + wid * 32;
  v8f z = {}; v8f acc0[6], acc1[6];
#pragma unroll
  for (int t = 0; t < 6; ++t) { acc0[t] = z; acc1[t] = z; }
  const _Float16* Arow0 = A + (size_t)(rowBase + lm) * 192;
  const _Float16* Arow1 = Arow0 + (size_t)16 * 192;
#pragma unroll
  for (int ks = 0; ks < 6; ++ks) {
    v16h a0 = ldA16(Arow0 + ks * 32, halfsel);
    v16h a1 = ldA16(Arow1 + ks * 32, halfsel);
#pragma unroll
    for (int nt = 0; nt < 6; ++nt) {
      v16h b = *(const v16h*)(Wt + (nt * 16 + lm) * 192 + ks * 32 + halfsel * 16);
      acc0[nt] = __builtin_amdgcn_wmma_f32_16x16x32_f16(
          false, a0, false, b, (short)0, acc0[nt], false, false);
      acc1[nt] = __builtin_amdgcn_wmma_f32_16x16x32_f16(
          false, a1, false, b, (short)0, acc1[nt], false, false);
    }
  }
#pragma unroll
  for (int nt = 0; nt < 6; ++nt) {
    int col = c0 + nt * 16 + lm;
    float bs = bias[col];
#pragma unroll
    for (int r = 0; r < 8; ++r) {
      size_t i0 = (size_t)(rowBase + halfsel * 8 + r) * 192 + col;
      size_t i1 = i0 + (size_t)16 * 192;
      x1[i0] = xres[i0] + acc0[nt][r] + bs;
      x1[i1] = xres[i1] + acc1[nt][r] + bs;
    }
  }
}

// ---------------------------------------------------------------------------
// fc1 + exact-erf GELU: y = gelu(h2 @ fc1_w + b)  -> f16 [TOK,768].
// ---------------------------------------------------------------------------
__global__ __launch_bounds__(256) void fc1_gemm_kernel(
    const _Float16* __restrict__ A, const float* __restrict__ W,
    const float* __restrict__ bias, _Float16* __restrict__ out) {
  __shared__ __align__(64) _Float16 Wt[96 * 192];
  int tid = threadIdx.x;
  int c0 = blockIdx.y * 96;
  for (int i = tid; i < 96 * 192; i += 256) {
    int n = i / 192, kk = i % 192;
    Wt[i] = (_Float16)W[kk * HID_ + c0 + n];
  }
  __syncthreads();
  int lane = tid & 31, wid = tid >> 5, halfsel = lane >> 4, lm = lane & 15;
  int rowBase = blockIdx.x * 256 + wid * 32;
  v8f z = {}; v8f acc0[6], acc1[6];
#pragma unroll
  for (int t = 0; t < 6; ++t) { acc0[t] = z; acc1[t] = z; }
  const _Float16* Arow0 = A + (size_t)(rowBase + lm) * 192;
  const _Float16* Arow1 = Arow0 + (size_t)16 * 192;
#pragma unroll
  for (int ks = 0; ks < 6; ++ks) {
    v16h a0 = ldA16(Arow0 + ks * 32, halfsel);
    v16h a1 = ldA16(Arow1 + ks * 32, halfsel);
#pragma unroll
    for (int nt = 0; nt < 6; ++nt) {
      v16h b = *(const v16h*)(Wt + (nt * 16 + lm) * 192 + ks * 32 + halfsel * 16);
      acc0[nt] = __builtin_amdgcn_wmma_f32_16x16x32_f16(
          false, a0, false, b, (short)0, acc0[nt], false, false);
      acc1[nt] = __builtin_amdgcn_wmma_f32_16x16x32_f16(
          false, a1, false, b, (short)0, acc1[nt], false, false);
    }
  }
#pragma unroll
  for (int nt = 0; nt < 6; ++nt) {
    int col = c0 + nt * 16 + lm;
    float bs = bias[col];
#pragma unroll
    for (int r = 0; r < 8; ++r) {
      int tok0 = rowBase + halfsel * 8 + r;
      float u0 = acc0[nt][r] + bs;
      float u1 = acc1[nt][r] + bs;
      float g0 = 0.5f * u0 * (1.0f + erff(u0 * 0.70710678118654752f));
      float g1 = 0.5f * u1 * (1.0f + erff(u1 * 0.70710678118654752f));
      out[(size_t)tok0 * HID_ + col]        = (_Float16)g0;
      out[(size_t)(tok0 + 16) * HID_ + col] = (_Float16)g1;
    }
  }
}

// ---------------------------------------------------------------------------
// fc2 + residual: out = x1 + y @ fc2_w + b (f32).  K=768 in 4 LDS chunks.
// ---------------------------------------------------------------------------
__global__ __launch_bounds__(256) void fc2_gemm_kernel(
    const _Float16* __restrict__ A, const float* __restrict__ W,
    const float* __restrict__ bias, const float* __restrict__ x1,
    float* __restrict__ out) {
  __shared__ __align__(64) _Float16 Wt[96 * 192];
  int tid = threadIdx.x;
  int c0 = blockIdx.y * 96;
  int lane = tid & 31, wid = tid >> 5, halfsel = lane >> 4, lm = lane & 15;
  int rowBase = blockIdx.x * 256 + wid * 32;
  v8f z = {}; v8f acc0[6], acc1[6];
#pragma unroll
  for (int t = 0; t < 6; ++t) { acc0[t] = z; acc1[t] = z; }
  const _Float16* Arow0 = A + (size_t)(rowBase + lm) * HID_;
  const _Float16* Arow1 = Arow0 + (size_t)16 * HID_;
  for (int kc = 0; kc < 4; ++kc) {
    __syncthreads();
    for (int i = tid; i < 96 * 192; i += 256) {
      int n = i / 192, kk = i % 192;
      Wt[i] = (_Float16)W[(size_t)(kc * 192 + kk) * 192 + c0 + n];
    }
    __syncthreads();
    if (kc < 3) {  // prefetch next A chunk into L2/L0 (global_prefetch_b8)
      __builtin_prefetch(Arow0 + (kc + 1) * 192, 0, 1);
      __builtin_prefetch(Arow1 + (kc + 1) * 192, 0, 1);
    }
#pragma unroll
    for (int ks = 0; ks < 6; ++ks) {
      v16h a0 = ldA16(Arow0 + kc * 192 + ks * 32, halfsel);
      v16h a1 = ldA16(Arow1 + kc * 192 + ks * 32, halfsel);
#pragma unroll
      for (int nt = 0; nt < 6; ++nt) {
        v16h b = *(const v16h*)(Wt + (nt * 16 + lm) * 192 + ks * 32 + halfsel * 16);
        acc0[nt] = __builtin_amdgcn_wmma_f32_16x16x32_f16(
            false, a0, false, b, (short)0, acc0[nt], false, false);
        acc1[nt] = __builtin_amdgcn_wmma_f32_16x16x32_f16(
            false, a1, false, b, (short)0, acc1[nt], false, false);
      }
    }
  }
#pragma unroll
  for (int nt = 0; nt < 6; ++nt) {
    int col = c0 + nt * 16 + lm;
    float bs = bias[col];
#pragma unroll
    for (int r = 0; r < 8; ++r) {
      size_t i0 = (size_t)(rowBase + halfsel * 8 + r) * 192 + col;
      size_t i1 = i0 + (size_t)16 * 192;
      out[i0] = x1[i0] + acc0[nt][r] + bs;
      out[i1] = x1[i1] + acc1[nt][r] + bs;
    }
  }
}

// ---------------------------------------------------------------------------
extern "C" void kernel_launch(void* const* d_in, const int* in_sizes, int n_in,
                              void* d_out, int out_size, void* d_ws, size_t ws_size,
                              hipStream_t stream) {
  (void)in_sizes; (void)n_in; (void)out_size; (void)ws_size;

  const float* x        = (const float*)d_in[0];
  // d_in[1] = TH (unused)
  const unsigned char* geo_mask = (const unsigned char*)d_in[2];
  const unsigned char* sem_mask = (const unsigned char*)d_in[3];
  const float* ln1_w = (const float*)d_in[4];
  const float* ln1_b = (const float*)d_in[5];
  const float* ln2_w = (const float*)d_in[6];
  const float* ln2_b = (const float*)d_in[7];
  // 8..25: {t,g,s} x {q,k,v} x {w,b}
  const float* proj_w = (const float*)d_in[26];
  const float* proj_b = (const float*)d_in[27];
  const float* fc1_w  = (const float*)d_in[28];
  const float* fc1_b  = (const float*)d_in[29];
  const float* fc2_w  = (const float*)d_in[30];
  const float* fc2_b  = (const float*)d_in[31];

  // workspace bump allocator (256B aligned)
  size_t off = 0;
  char* base = (char*)d_ws;
  auto alloc = [&](size_t bytes) -> char* {
    char* p = base + off;
    off += (bytes + 255) & ~(size_t)255;
    return p;
  };
  _Float16* h16      = (_Float16*)alloc((size_t)TOK * D_   * 2);
  _Float16* qkv[9];
  for (int i = 0; i < 9; ++i) qkv[i] = (_Float16*)alloc((size_t)TOK * BR_ * 2);
  _Float16* attn_cat = (_Float16*)alloc((size_t)TOK * D_   * 2);
  float*    x1       = (float*)   alloc((size_t)TOK * D_   * 4);
  _Float16* h2       = (_Float16*)alloc((size_t)TOK * D_   * 2);
  _Float16* y16      = (_Float16*)alloc((size_t)TOK * HID_ * 2);

  // 1) LN1 -> f16
  ln_f16_kernel<<<TOK / 8, 256, 0, stream>>>(x, ln1_w, ln1_b, h16);

  // 2) nine QKV projections (tq,tk,tv,gq,gk,gv,sq,sk,sv)
  for (int i = 0; i < 9; ++i) {
    qkv_gemm_kernel<<<TOK / 256, 256, 0, stream>>>(
        h16, (const float*)d_in[8 + 2 * i], (const float*)d_in[9 + 2 * i], qkv[i]);
  }

  // 3) temporal branch -> attn_cat[:, 0:64]
  temporal_attn_kernel<<<B_ * N_, 32, 0, stream>>>(qkv[0], qkv[1], qkv[2], attn_cat);

  // 4) spatial branches -> attn_cat[:, 64:128] and [:, 128:192]
  dim3 sg(N_ / 128, B_ * T_, H_);
  spatial_attn_kernel<<<sg, 256, 0, stream>>>(qkv[3], qkv[4], qkv[5], geo_mask, attn_cat, 64);
  spatial_attn_kernel<<<sg, 256, 0, stream>>>(qkv[6], qkv[7], qkv[8], sem_mask, attn_cat, 128);

  // 5) output projection + residual -> x1 (f32)
  proj_gemm_kernel<<<dim3(TOK / 256, 2), 256, 0, stream>>>(attn_cat, proj_w, proj_b, x, x1);

  // 6) LN2 -> f16
  ln_f16_kernel<<<TOK / 8, 256, 0, stream>>>(x1, ln2_w, ln2_b, h2);

  // 7) fc1 + exact GELU -> f16 [TOK,768]
  fc1_gemm_kernel<<<dim3(TOK / 256, HID_ / 96), 256, 0, stream>>>(h2, fc1_w, fc1_b, y16);

  // 8) fc2 + residual -> d_out (f32)
  fc2_gemm_kernel<<<dim3(TOK / 256, 2), 256, 0, stream>>>(y16, fc2_w, fc2_b, x1, (float*)d_out);
}
